// VectorQuantizer_30837865185315
// MI455X (gfx1250) — matrix operand order repaired and checked by reading, not compiled, not run on hardware
//
#include <hip/hip_runtime.h>
#include <hip/hip_bf16.h>

// ---------------------------------------------------------------------------
// VQ-VAE vector quantizer for MI455X (gfx1250, wave32, WMMA).
//
// distances d[n,k] = |a|^2 - 2 a.b + |b|^2.  The -2 a.b term is a
// [262144 x 64] x [64 x 512] GEMM done with V_WMMA_F32_16X16X32_BF16 using a
// hi/lo bf16 split of both operands (3 WMMAs per K-step) for ~fp32 accuracy
// at bf16-WMMA throughput.  |a|^2 / |b|^2 are exact fp32.  Codebook (hi+lo
// bf16) + z_e tile are staged in ~219 KB of LDS (CDNA5 WGP has 320 KB).
// Inner loop: 2 code blocks / iteration, 4 independent WMMA chains, all
// 16 ds_load_b128 issued before the WMMA block (partial dscnt waits).
// LDS caps us at 1 WG/WGP (= 4 waves/SIMD), so launch_bounds(512,4) frees
// the register allocator to keep all fragments live.
// ---------------------------------------------------------------------------

typedef __attribute__((ext_vector_type(16))) __bf16 v16bf;
typedef __attribute__((ext_vector_type(8)))  __bf16 v8bf;
typedef __attribute__((ext_vector_type(8)))  float  v8f;

#define KCODES   512
#define DIM      64
#define TM       256          // rows per workgroup
#define NTHREADS 512          // 16 waves
#define LSTRIDE  72           // bf16 elems per LDS row (64 + 8 pad -> no bank conflicts)

// LDS byte offsets (base is 16B aligned; all offsets multiples of 16)
#define OFF_BH   0                               // codebook hi : 512*72*2 = 73728
#define OFF_BL   73728                           // codebook lo : 73728
#define OFF_AH   147456                          // tile hi     : 256*72*2 = 36864
#define OFF_AL   184320                          // tile lo     : 36864
#define OFF_BSQ  221184                          // 512 f32
#define OFF_ASQ  223232                          // 256 f32
#define LDS_BYTES 224256

__device__ __forceinline__ unsigned short bf16_rne(float f) {
    unsigned int u = __float_as_uint(f);
    return (unsigned short)((u + 0x7FFFu + ((u >> 16) & 1u)) >> 16);
}
__device__ __forceinline__ float bf16_to_f(unsigned short h) {
    return __uint_as_float(((unsigned int)h) << 16);
}
// build a v16bf WMMA fragment from two 16B LDS chunks
__device__ __forceinline__ v16bf ldfrag(const unsigned short* p0,
                                        const unsigned short* p1) {
    v8bf x0 = *(const v8bf*)p0;
    v8bf x1 = *(const v8bf*)p1;
    return __builtin_shufflevector(x0, x1,
        0,1,2,3,4,5,6,7,8,9,10,11,12,13,14,15);
}
#define WMMA_BF16(A, B, C) \
    __builtin_amdgcn_wmma_f32_16x16x32_bf16(false, (A), false, (B), (short)0, (C), false, false)

__global__ void vq_zero_kernel(float* ws, int n) {
    int t = blockIdx.x * blockDim.x + threadIdx.x;
    if (t < n) ws[t] = 0.0f;
}

__global__ __launch_bounds__(NTHREADS, 4)   // 4 waves/SIMD floor -> ~256 VGPR budget
void vq_main_kernel(const float* __restrict__ z_e, const float* __restrict__ cb,
                    float* __restrict__ o_zq, float* __restrict__ o_idx,
                    float* __restrict__ o_mind,
                    float* __restrict__ ws_cnt, float* __restrict__ ws_sum) {
    extern __shared__ char smem[];
    unsigned short* sBh = (unsigned short*)(smem + OFF_BH);
    unsigned short* sBl = (unsigned short*)(smem + OFF_BL);
    unsigned short* sAh = (unsigned short*)(smem + OFF_AH);
    unsigned short* sAl = (unsigned short*)(smem + OFF_AL);
    float* sBsq = (float*)(smem + OFF_BSQ);
    float* sAsq = (float*)(smem + OFF_ASQ);

    const int tid = threadIdx.x;
    const size_t rowBase = (size_t)blockIdx.x * TM;

    // ---- stage codebook -> LDS bf16 hi/lo (coalesced float4 stream) ----
    const float4* cb4 = (const float4*)cb;                 // 512*16 float4
    for (int f = tid; f < KCODES * (DIM / 4); f += NTHREADS) {
        int code = f >> 4, k4 = (f & 15) << 2;
        float4 v = cb4[f];
        unsigned short h0 = bf16_rne(v.x), h1 = bf16_rne(v.y),
                       h2 = bf16_rne(v.z), h3 = bf16_rne(v.w);
        unsigned short l0 = bf16_rne(v.x - bf16_to_f(h0)),
                       l1 = bf16_rne(v.y - bf16_to_f(h1)),
                       l2 = bf16_rne(v.z - bf16_to_f(h2)),
                       l3 = bf16_rne(v.w - bf16_to_f(h3));
        *(ushort4*)(sBh + code * LSTRIDE + k4) = make_ushort4(h0, h1, h2, h3);
        *(ushort4*)(sBl + code * LSTRIDE + k4) = make_ushort4(l0, l1, l2, l3);
    }
    // ---- stage z_e tile -> LDS bf16 hi/lo ----
    const float4* ze4 = (const float4*)z_e;
    for (int f = tid; f < TM * (DIM / 4); f += NTHREADS) {
        int r = f >> 4, k4 = (f & 15) << 2;
        float4 v = ze4[rowBase * (DIM / 4) + f];
        unsigned short h0 = bf16_rne(v.x), h1 = bf16_rne(v.y),
                       h2 = bf16_rne(v.z), h3 = bf16_rne(v.w);
        unsigned short l0 = bf16_rne(v.x - bf16_to_f(h0)),
                       l1 = bf16_rne(v.y - bf16_to_f(h1)),
                       l2 = bf16_rne(v.z - bf16_to_f(h2)),
                       l3 = bf16_rne(v.w - bf16_to_f(h3));
        *(ushort4*)(sAh + r * LSTRIDE + k4) = make_ushort4(h0, h1, h2, h3);
        *(ushort4*)(sAl + r * LSTRIDE + k4) = make_ushort4(l0, l1, l2, l3);
    }
    // ---- exact fp32 |b|^2 and |a|^2 (rows just touched, cache hot) ----
    if (tid < KCODES) {
        const float* p = cb + (size_t)tid * DIM;
        float s = 0.0f;
        #pragma unroll 8
        for (int d = 0; d < DIM; ++d) s += p[d] * p[d];
        sBsq[tid] = s;
    }
    if (tid < TM) {
        const float* p = z_e + (rowBase + tid) * DIM;
        float s = 0.0f;
        #pragma unroll 8
        for (int d = 0; d < DIM; ++d) s += p[d] * p[d];
        sAsq[tid] = s;
    }
    __syncthreads();

    const int lane = tid & 31;
    const int wave = tid >> 5;         // 0..15 -> M-tile rows [wave*16, wave*16+16)
    const int hv   = lane >> 4;        // half-wave
    const int lh   = lane & 15;        // lane in half = column / A-row

    // ---- A fragments (ISA 16-bit A layout): row = wave*16+lh,
    //      lanes 0-15: K c0..c0+7 & c0+16..c0+23; lanes 16-31: +8 ----
    const int mrow = wave * 16 + lh;
    v16bf ah[2], al[2];
    #pragma unroll
    for (int ks = 0; ks < 2; ++ks) {
        const unsigned short* pa = sAh + mrow * LSTRIDE + ks * 32 + hv * 8;
        const unsigned short* pb = sAl + mrow * LSTRIDE + ks * 32 + hv * 8;
        ah[ks] = ldfrag(pa, pa + 16);
        al[ks] = ldfrag(pb, pb + 16);
    }

    float minv[8];
    int   mini[8];
    #pragma unroll
    for (int i = 0; i < 8; ++i) { minv[i] = 3.4e38f; mini[i] = 0; }

    const v8f z8 = {0.f, 0.f, 0.f, 0.f, 0.f, 0.f, 0.f, 0.f};

    // ---- sweep 512 codes, 2x16-column blocks / iteration ----
    for (int cbk = 0; cbk < KCODES / 16; cbk += 2) {
        const int code0 = cbk * 16 + lh;            // B column owned by this lane
        const int code1 = code0 + 16;

        // issue ALL 16 ds_load_b128 (B frags) + 2 b^2 scalars, then waits
        v16bf Bh0[2], Bl0[2], Bh1[2], Bl1[2];
        #pragma unroll
        for (int ks = 0; ks < 2; ++ks) {
            // B layout (matches A): lanes 0-15 hold K=ks*32+0..7 & +16..23, lanes 16-31 +8
            const int c0 = ks * 32 + hv * 8;
            const unsigned short* p;
            p = sBh + code0 * LSTRIDE + c0;  Bh0[ks] = ldfrag(p, p + 16);
            p = sBl + code0 * LSTRIDE + c0;  Bl0[ks] = ldfrag(p, p + 16);
            p = sBh + code1 * LSTRIDE + c0;  Bh1[ks] = ldfrag(p, p + 16);
            p = sBl + code1 * LSTRIDE + c0;  Bl1[ks] = ldfrag(p, p + 16);
        }
        const float bq0 = sBsq[code0];
        const float bq1 = sBsq[code1];

        // 4 independent accumulator chains (block x K-half), 3 WMMAs each;
        // first WMMA of each chain uses the inline-0 C operand.  Bl-consuming
        // WMMAs come last so their loads have the longest time to land.
        v8f a00, a01, a10, a11;
        a00 = WMMA_BF16(ah[0], Bh0[0], z8);
        a01 = WMMA_BF16(ah[1], Bh0[1], z8);
        a10 = WMMA_BF16(ah[0], Bh1[0], z8);
        a11 = WMMA_BF16(ah[1], Bh1[1], z8);
        a00 = WMMA_BF16(al[0], Bh0[0], a00);
        a01 = WMMA_BF16(al[1], Bh0[1], a01);
        a10 = WMMA_BF16(al[0], Bh1[0], a10);
        a11 = WMMA_BF16(al[1], Bh1[1], a11);
        a00 = WMMA_BF16(ah[0], Bl0[0], a00);
        a01 = WMMA_BF16(ah[1], Bl0[1], a01);
        a10 = WMMA_BF16(ah[0], Bl1[0], a10);
        a11 = WMMA_BF16(ah[1], Bl1[1], a11);

        // acc[i] belongs to row (wave*16 + hv*8 + i); check code0 before
        // code1 with strict < to keep jnp.argmin first-index semantics.
        #pragma unroll
        for (int i = 0; i < 8; ++i) {
            float d0 = bq0 - 2.0f * (a00[i] + a01[i]);
            if (d0 < minv[i]) { minv[i] = d0; mini[i] = code0; }
            float d1 = bq1 - 2.0f * (a10[i] + a11[i]);
            if (d1 < minv[i]) { minv[i] = d1; mini[i] = code1; }
        }
    }

    // ---- argmin reduction across the 16 lanes of each half (first-index ties) ----
    #pragma unroll
    for (int mask = 1; mask < 16; mask <<= 1) {
        #pragma unroll
        for (int i = 0; i < 8; ++i) {
            float ov = __shfl_xor(minv[i], mask, 32);
            int   oi = __shfl_xor(mini[i], mask, 32);
            if (ov < minv[i] || (ov == minv[i] && oi < mini[i])) {
                minv[i] = ov; mini[i] = oi;
            }
        }
    }

    const int    rloc  = wave * 16 + hv * 8;        // local row base for this half
    const size_t rglob = rowBase + rloc;

    if (lh == 0) {
        #pragma unroll
        for (int i = 0; i < 8; ++i) {
            o_idx[rglob + i]  = (float)mini[i];
            o_mind[rglob + i] = sAsq[rloc + i] + minv[i];
            atomicAdd(ws_cnt + mini[i], 1.0f);      // cluster-size histogram
        }
    }
    // z_q gather (fp32 codebook, L2-resident) + EMA embedding scatter-add
    #pragma unroll
    for (int i = 0; i < 8; ++i) {
        const int    code = mini[i];
        const size_t r    = rglob + i;
        float4 c4 = ((const float4*)cb)[code * (DIM / 4) + lh];
        ((float4*)o_zq)[r * (DIM / 4) + lh] = c4;   // z_q_st == z_q numerically
        float4 z4 = ((const float4*)z_e)[r * (DIM / 4) + lh];
        float* dst = ws_sum + (size_t)code * DIM + lh * 4;
        atomicAdd(dst + 0, z4.x);
        atomicAdd(dst + 1, z4.y);
        atomicAdd(dst + 2, z4.z);
        atomicAdd(dst + 3, z4.w);
    }
}

__global__ void vq_finalize_kernel(const float* __restrict__ ema_cs,
                                   const float* __restrict__ ema_es,
                                   const float* __restrict__ ws_cnt,
                                   const float* __restrict__ ws_sum,
                                   float* __restrict__ o_cb,
                                   float* __restrict__ o_cs,
                                   float* __restrict__ o_es) {
    int t = blockIdx.x * blockDim.x + threadIdx.x;  // < K*D
    int k = t >> 6;
    float ncs = 0.99f * ema_cs[k] + 0.01f * ws_cnt[k];
    float nes = 0.99f * ema_es[t] + 0.01f * ws_sum[t];
    o_es[t] = nes;
    o_cb[t] = nes / (ncs + 1e-5f);
    if ((t & 63) == 0) o_cs[k] = ncs;
}

extern "C" void kernel_launch(void* const* d_in, const int* in_sizes, int n_in,
                              void* d_out, int out_size, void* d_ws, size_t ws_size,
                              hipStream_t stream) {
    const float* z_e    = (const float*)d_in[0];   // [512,512,64]
    const float* cb     = (const float*)d_in[1];   // [512,64]
    const float* ema_cs = (const float*)d_in[2];   // [512]
    const float* ema_es = (const float*)d_in[3];   // [512,64]

    float* out    = (float*)d_out;
    float* o_zq   = out;                            // 16777216
    float* o_idx  = out + 16777216;                 // 262144
    float* o_mind = out + 17039360;                 // 262144
    float* o_cb   = out + 17301504;                 // 32768
    float* o_cs   = out + 17334272;                 // 512
    float* o_es   = out + 17334784;                 // 32768

    float* wsf    = (float*)d_ws;
    float* ws_cnt = wsf;                            // [512]
    float* ws_sum = wsf + 512;                      // [512*64]

    (void)in_sizes; (void)n_in; (void)out_size; (void)ws_size;

    // allow > 64 KB dynamic LDS (CDNA5 WGP has 320 KB)
    (void)hipFuncSetAttribute((const void*)vq_main_kernel,
                              hipFuncAttributeMaxDynamicSharedMemorySize, LDS_BYTES);

    const int nacc = KCODES + KCODES * DIM;         // 33280 accumulator floats
    vq_zero_kernel<<<(nacc + 255) / 256, 256, 0, stream>>>(wsf, nacc);
    vq_main_kernel<<<dim3(1024), dim3(NTHREADS), LDS_BYTES, stream>>>(
        z_e, cb, o_zq, o_idx, o_mind, ws_cnt, ws_sum);
    vq_finalize_kernel<<<dim3(128), dim3(256), 0, stream>>>(
        ema_cs, ema_es, ws_cnt, ws_sum, o_cb, o_cs, o_es);
}
